// Decoder_30210799960238
// MI455X (gfx1250) — compile-verified, matrix operand order checked by reading, not run on hardware
//
#include <hip/hip_runtime.h>
#include <hip/hip_bf16.h>

typedef __attribute__((ext_vector_type(16))) __bf16 v16bf;
typedef __attribute__((ext_vector_type(8)))  float  v8f;
typedef __attribute__((ext_vector_type(4)))  int    i4v;

#define MBLK 64
#define NBLK 128
#define KBLK 64

#if __has_builtin(__builtin_amdgcn_global_load_async_to_lds_b128)
#define HAVE_ASYNC_LDS 1
typedef __attribute__((address_space(1))) i4v glob_i4_t;
typedef __attribute__((address_space(3))) i4v lds_i4_t;
#endif

__device__ __forceinline__ void wait_async_lds() {
#ifdef HAVE_ASYNC_LDS
#if __has_builtin(__builtin_amdgcn_s_wait_asynccnt)
    __builtin_amdgcn_s_wait_asynccnt(0);
#else
    asm volatile("s_wait_asynccnt 0x0" ::: "memory");
#endif
#endif
}

__device__ __forceinline__ unsigned short f2bf(float f) {
    unsigned int u = __float_as_uint(f);
    u += 0x7FFFu + ((u >> 16) & 1u);
    return (unsigned short)(u >> 16);
}

__device__ __forceinline__ unsigned int hashrng(unsigned int a, unsigned int b) {
    unsigned int x = a * 0x9E3779B1u ^ (b + 0x85EBCA77u) * 0xC2B2AE3Du;
    x ^= x >> 16; x *= 0x7FEB352Du; x ^= x >> 15; x *= 0x846CA68Bu; x ^= x >> 16;
    return x;
}

// Fragment gather per ISA 7.12.2: lanes 0-15 hold K{0..7,16..23}, 16-31 K{8..15,24..31}
__device__ __forceinline__ v16bf load_frag(const __bf16* p) {
    v16bf f;
#pragma unroll
    for (int e = 0; e < 16; ++e) f[e] = p[(e & 7) + 16 * (e >> 3)];
    return f;
}

// ---------------------------------------------------------------------------
// bf16 GEMM, C[M,N] = A[M,K] * B[N,K]^T (+bias[N]), f32 accumulate.
// 64x128 tile/WG, 8 waves x (16x64), double-buffered LDS, async global->LDS
// when available, split-K over blockIdx.z writing partials at C + z*Czstride.
// M mult of 64, N of 128, Kchunk of 64.
// ---------------------------------------------------------------------------
__global__ __launch_bounds__(256) void gemm_bf16_nt(
    const unsigned short* __restrict__ A, int lda,
    const unsigned short* __restrict__ B, int ldb,
    float* __restrict__ C, int ldc, int Kchunk,
    const float* __restrict__ bias, size_t Czstride)
{
    __shared__ __align__(16) unsigned short sA[2][MBLK * KBLK];
    __shared__ __align__(16) unsigned short sB[2][NBLK * KBLK];

    const int tid   = threadIdx.x;
    const int lane  = tid & 31;
    const int wave  = tid >> 5;
    const int waveM = wave & 3;   // 0..3 -> M sub-tile
    const int waveN = wave >> 2;  // 0..1 -> N half
    const int Mbase = blockIdx.x * MBLK;
    const int Nbase = blockIdx.y * NBLK;
    const int kstart = blockIdx.z * Kchunk;
    C += (size_t)blockIdx.z * Czstride;

    const int r     = lane & 15;
    const int khalf = lane >> 4;

    v8f acc[4];
#pragma unroll
    for (int j = 0; j < 4; ++j)
#pragma unroll
        for (int e = 0; e < 8; ++e) acc[j][e] = 0.0f;

    auto stage = [&](int kb, int buf) {
        // A tile: 64x64 bf16 = 512 x b128
        for (int i = tid; i < (MBLK * KBLK) / 8; i += 256) {
            int row = i >> 3, col8 = i & 7;  // KBLK/8 == 8
            const unsigned short* gp =
                A + (size_t)(Mbase + row) * lda + kb + col8 * 8;
#ifdef HAVE_ASYNC_LDS
            __builtin_amdgcn_global_load_async_to_lds_b128(
                (glob_i4_t*)gp, (lds_i4_t*)&sA[buf][i * 8], 0, 0);
#else
            ((uint4*)sA[buf])[i] = *(const uint4*)gp;
#endif
        }
        // B tile: 128x64 bf16 = 1024 x b128
        for (int i = tid; i < (NBLK * KBLK) / 8; i += 256) {
            int row = i >> 3, col8 = i & 7;
            const unsigned short* gp =
                B + (size_t)(Nbase + row) * ldb + kb + col8 * 8;
#ifdef HAVE_ASYNC_LDS
            __builtin_amdgcn_global_load_async_to_lds_b128(
                (glob_i4_t*)gp, (lds_i4_t*)&sB[buf][i * 8], 0, 0);
#else
            ((uint4*)sB[buf])[i] = *(const uint4*)gp;
#endif
        }
    };

    const int nk = Kchunk / KBLK;
    stage(kstart, 0);
    wait_async_lds();
    __syncthreads();

    for (int i = 0; i < nk; ++i) {
        if (i + 1 < nk) stage(kstart + (i + 1) * KBLK, (i + 1) & 1);

        const __bf16* lA = (const __bf16*)sA[i & 1];
        const __bf16* lB = (const __bf16*)sB[i & 1];
#pragma unroll
        for (int ks = 0; ks < KBLK; ks += 32) {
            // batch all fragment loads before the WMMA burst
            v16bf a = load_frag(lA + (waveM * 16 + r) * KBLK + ks + 8 * khalf);
            const __bf16* pb0 = lB + (waveN * 64 + r) * KBLK + ks + 8 * khalf;
            v16bf b0 = load_frag(pb0);
            v16bf b1 = load_frag(pb0 + 16 * KBLK);
            v16bf b2 = load_frag(pb0 + 32 * KBLK);
            v16bf b3 = load_frag(pb0 + 48 * KBLK);
            acc[0] = __builtin_amdgcn_wmma_f32_16x16x32_bf16(
                false, a, false, b0, (short)0, acc[0], false, false);
            acc[1] = __builtin_amdgcn_wmma_f32_16x16x32_bf16(
                false, a, false, b1, (short)0, acc[1], false, false);
            acc[2] = __builtin_amdgcn_wmma_f32_16x16x32_bf16(
                false, a, false, b2, (short)0, acc[2], false, false);
            acc[3] = __builtin_amdgcn_wmma_f32_16x16x32_bf16(
                false, a, false, b3, (short)0, acc[3], false, false);
        }
        wait_async_lds();   // next buffer's async loads done (overlapped with WMMA)
        __syncthreads();
    }

    // D layout: VGPR e, lanes 0-15 -> M=e, N=lane; lanes 16-31 -> M=e+8, N=lane-16
#pragma unroll
    for (int j = 0; j < 4; ++j) {
#pragma unroll
        for (int e = 0; e < 8; ++e) {
            int m    = e + 8 * khalf;
            int gcol = Nbase + waveN * 64 + j * 16 + r;
            int grow = Mbase + waveM * 16 + m;
            float v  = acc[j][e];
            if (bias) v += bias[gcol];
            C[(size_t)grow * ldc + gcol] = v;
        }
    }
}

// ---------------------------------------------------------------------------
// Weight / input preparation kernels
// ---------------------------------------------------------------------------
__global__ void cvt_wstack_kernel(const float* __restrict__ w_ih,
                                  const float* __restrict__ w_hh,
                                  unsigned short* __restrict__ Wstack)
{
    size_t idx = (size_t)blockIdx.x * 256 + threadIdx.x;
    if (idx >= (size_t)4096 * 2048) return;
    int n = (int)(idx >> 11);
    int k = (int)(idx & 2047);
    float v = (k < 1024) ? w_ih[(size_t)n * 1024 + k]
                         : w_hh[(size_t)n * 1024 + (k - 1024)];
    Wstack[idx] = f2bf(v);
}

__global__ void cvt_bf_kernel(const float* __restrict__ src,
                              unsigned short* __restrict__ dst, size_t n)
{
    size_t idx = (size_t)blockIdx.x * 256 + threadIdx.x;
    if (idx < n) dst[idx] = f2bf(src[idx]);
}

__global__ void bias_kernel(const float* __restrict__ bih,
                            const float* __restrict__ bhh,
                            float* __restrict__ bs)
{
    int idx = blockIdx.x * 256 + threadIdx.x;
    if (idx < 4096) bs[idx] = bih[idx] + bhh[idx];
}

// Fold loc_dense[128,32] @ loc_conv[32,31] -> combined filter Cf[128,32] (k=31 pad 0)
__global__ void cfilt_kernel(const float* __restrict__ loc_dense,
                             const float* __restrict__ loc_conv,
                             float* __restrict__ Cf)
{
    int idx = blockIdx.x * 256 + threadIdx.x;
    if (idx >= 128 * 32) return;
    int a = idx >> 5, k = idx & 31;
    float s = 0.0f;
    if (k < 31)
        for (int f = 0; f < 32; ++f)
            s += loc_dense[a * 32 + f] * loc_conv[f * 31 + k];
    Cf[idx] = s;
}

__global__ void init_state_kernel(const float* __restrict__ spk,
                                  unsigned short* __restrict__ Xbf,
                                  float* __restrict__ h, float* __restrict__ c)
{
    int idx = blockIdx.x * 256 + threadIdx.x;
    if (idx < 64 * 2048) {
        int b = idx >> 11, col = idx & 2047;
        unsigned short v = 0;
        if (col >= 768 && col < 1024) v = f2bf(spk[b * 256 + (col - 768)]);
        Xbf[idx] = v;
    }
    if (idx < 64 * 1024) { h[idx] = 0.0f; c[idx] = 0.0f; }
}

// ---------------------------------------------------------------------------
// Per-step kernels
// ---------------------------------------------------------------------------
// PreNet: teacher[:,t-1] -> 256 -> 256 with relu + dropout(0.5), into Xbf[:,0:256]
__global__ __launch_bounds__(256) void prenet_kernel(
    const float* __restrict__ teacher,
    const float* __restrict__ p1w, const float* __restrict__ p1b,
    const float* __restrict__ p2w, const float* __restrict__ p2b,
    unsigned short* __restrict__ Xbf, int t)
{
    int b = blockIdx.x, tid = threadIdx.x;
    __shared__ float mel[80];
    __shared__ float h1[256];
    if (tid < 80)
        mel[tid] = (t == 0) ? 0.0f
                            : teacher[((size_t)b * 600 + (t - 1)) * 80 + tid];
    __syncthreads();
    {
        float s = p1b[tid];
        const float* w = p1w + tid * 80;
        for (int k = 0; k < 80; ++k) s += w[k] * mel[k];
        s = fmaxf(s, 0.0f);
        s = (hashrng(2u * t + 0u, (unsigned)(b * 256 + tid)) & 1u) ? s * 2.0f : 0.0f;
        h1[tid] = s;
    }
    __syncthreads();
    {
        float s = p2b[tid];
        const float* w = p2w + tid * 256;
        for (int k = 0; k < 256; ++k) s += w[k] * h1[k];
        s = fmaxf(s, 0.0f);
        s = (hashrng(2u * t + 1u, (unsigned)(b * 256 + tid)) & 1u) ? s * 2.0f : 0.0f;
        Xbf[(size_t)b * 2048 + tid] = f2bf(s);
    }
}

// Location-sensitive attention scores (conv folded into Cf[128,31]); pq from 4 split-K partials
__global__ __launch_bounds__(128) void score_kernel(
    const float* __restrict__ projkey,   // [64,400,128]
    const float* __restrict__ pqP,       // [4][64,128] partials
    const float* __restrict__ Cf,        // [128,32]
    const float* __restrict__ v,         // [128]
    const int*   __restrict__ mask,      // [64,400]
    const float* __restrict__ attn_hist, // d_out attns [64,600,400]
    float* __restrict__ scores, int t)
{
    int b = blockIdx.x, tid = threadIdx.x;
    int l0 = blockIdx.y * 128;
    __shared__ float win[128 + 30];
    __shared__ float sC[128 * 31];
    __shared__ float spq[128];
    __shared__ float sv[128];
    for (int i = tid; i < 158; i += 128) {
        int l = l0 + i - 15;
        float w = 0.0f;
        if (t > 0 && l >= 0 && l < 400)
            w = attn_hist[((size_t)b * 600 + (t - 1)) * 400 + l];
        win[i] = w;
    }
    for (int i = tid; i < 128 * 31; i += 128) {
        int a = i / 31, k = i % 31;
        sC[i] = Cf[a * 32 + k];
    }
    {
        int base = b * 128 + tid;
        spq[tid] = pqP[base] + pqP[8192 + base] + pqP[16384 + base] + pqP[24576 + base];
        sv[tid]  = v[tid];
    }
    __syncthreads();
    int l = l0 + tid;
    if (l < 400) {
        const float* pk = projkey + ((size_t)b * 400 + l) * 128;
        float s = 0.0f;
        for (int a = 0; a < 128; ++a) {
            float pl = 0.0f;
            const float* cf = sC + a * 31;
#pragma unroll
            for (int k = 0; k < 31; ++k) pl += cf[k] * win[tid + k];
            s += sv[a] * tanhf(pk[a] + spq[a] + pl);
        }
        if (mask[b * 400 + l] == 0) s -= 1e9f;
        scores[b * 400 + l] = s;
    }
}

// Softmax over L=400 + context = attn @ enc, write attn to output history
__global__ __launch_bounds__(256) void softmax_ctx_kernel(
    const float* __restrict__ scores,
    const float* __restrict__ enc,       // [64,400,512] f32
    float* __restrict__ attn_out,        // d_out attns [64,600,400]
    float* __restrict__ ctx,             // [64,512]
    unsigned short* __restrict__ Xbf, int t)
{
    int b = blockIdx.x, tid = threadIdx.x;
    __shared__ float sa[400];
    __shared__ float red[256];
    float m = -1e30f;
    for (int l = tid; l < 400; l += 256) {
        float x = scores[b * 400 + l];
        sa[l] = x;
        m = fmaxf(m, x);
    }
    red[tid] = m;
    __syncthreads();
    for (int o = 128; o > 0; o >>= 1) {
        if (tid < o) red[tid] = fmaxf(red[tid], red[tid + o]);
        __syncthreads();
    }
    m = red[0];
    __syncthreads();
    float sum = 0.0f;
    for (int l = tid; l < 400; l += 256) {
        float e = __expf(sa[l] - m);
        sa[l] = e;
        sum += e;
    }
    red[tid] = sum;
    __syncthreads();
    for (int o = 128; o > 0; o >>= 1) {
        if (tid < o) red[tid] += red[tid + o];
        __syncthreads();
    }
    float inv = 1.0f / red[0];
    __syncthreads();
    for (int l = tid; l < 400; l += 256) {
        sa[l] *= inv;
        attn_out[((size_t)b * 600 + t) * 400 + l] = sa[l];
    }
    __syncthreads();
    for (int d = tid; d < 512; d += 256) {
        float s = 0.0f;
        const float* e = enc + (size_t)b * 400 * 512 + d;
        for (int l = 0; l < 400; ++l) s += sa[l] * e[(size_t)l * 512];
        ctx[b * 512 + d] = s;
        Xbf[(size_t)b * 2048 + 256 + d] = f2bf(s);
    }
}

// LSTM elementwise: reduce 4 split-K gate partials + bias -> h,c; h also to Xbf bf16
__global__ __launch_bounds__(256) void lstm_kernel(
    const float* __restrict__ gatesP,    // [4][64,4096]
    const float* __restrict__ bs,        // [4096] combined bias
    float* __restrict__ h, float* __restrict__ c,
    unsigned short* __restrict__ Xbf)
{
    const size_t S = (size_t)64 * 4096;
    int idx = blockIdx.x * 256 + threadIdx.x;  // 0..65535
    int b = idx >> 10, n = idx & 1023;
    size_t g0 = (size_t)b * 4096;
    float ig = bs[n],        fg = bs[1024 + n];
    float gg = bs[2048 + n], og = bs[3072 + n];
#pragma unroll
    for (int z = 0; z < 4; ++z) {
        const float* g = gatesP + z * S + g0;
        ig += g[n]; fg += g[1024 + n]; gg += g[2048 + n]; og += g[3072 + n];
    }
    float si = 1.0f / (1.0f + __expf(-ig));
    float sf = 1.0f / (1.0f + __expf(-fg));
    float so = 1.0f / (1.0f + __expf(-og));
    float c2 = sf * c[idx] + si * tanhf(gg);
    float h2 = so * tanhf(c2);
    c[idx] = c2;
    h[idx] = h2;
    Xbf[(size_t)b * 2048 + 1024 + n] = f2bf(h2);
}

// mel + stop projections from co=[h,ctx,spk] (1792)
__global__ __launch_bounds__(256) void output_kernel(
    const float* __restrict__ h, const float* __restrict__ ctx,
    const float* __restrict__ spk,
    const float* __restrict__ lin_w, const float* __restrict__ lin_b,
    const float* __restrict__ stop_w, const float* __restrict__ stop_b,
    float* __restrict__ mels, float* __restrict__ stops, int t)
{
    int b = blockIdx.x, tid = threadIdx.x;
    __shared__ float co[1792];
    for (int i = tid; i < 1024; i += 256) co[i] = h[b * 1024 + i];
    for (int i = tid; i < 512; i += 256) co[1024 + i] = ctx[b * 512 + i];
    for (int i = tid; i < 256; i += 256) co[1536 + i] = spk[b * 256 + i];
    __syncthreads();
    if (tid < 80) {
        float s = lin_b[tid];
        const float* w = lin_w + tid * 1792;
        for (int k = 0; k < 1792; ++k) s += w[k] * co[k];
        mels[((size_t)b * 600 + t) * 80 + tid] = s;
    } else if (tid == 80) {
        float s = stop_b[0];
        for (int k = 0; k < 1792; ++k) s += stop_w[k] * co[k];
        stops[(size_t)b * 600 + t] = s;
    }
}

// ---------------------------------------------------------------------------
extern "C" void kernel_launch(void* const* d_in, const int* in_sizes, int n_in,
                              void* d_out, int out_size, void* d_ws, size_t ws_size,
                              hipStream_t stream)
{
    (void)in_sizes; (void)n_in; (void)out_size; (void)ws_size;
    const float* enc      = (const float*)d_in[0];
    const int*   mask     = (const int*)d_in[1];
    const float* spk      = (const float*)d_in[2];
    const float* teacher  = (const float*)d_in[3];
    const float* w1       = (const float*)d_in[4];
    const float* w2       = (const float*)d_in[5];
    const float* v        = (const float*)d_in[6];
    const float* loc_conv = (const float*)d_in[7];
    const float* loc_dense= (const float*)d_in[8];
    const float* p1w      = (const float*)d_in[9];
    const float* p1b      = (const float*)d_in[10];
    const float* p2w      = (const float*)d_in[11];
    const float* p2b      = (const float*)d_in[12];
    const float* w_ih     = (const float*)d_in[13];
    const float* w_hh     = (const float*)d_in[14];
    const float* b_ih     = (const float*)d_in[15];
    const float* b_hh     = (const float*)d_in[16];
    const float* lin_w    = (const float*)d_in[17];
    const float* lin_b    = (const float*)d_in[18];
    const float* stop_w   = (const float*)d_in[19];
    const float* stop_b   = (const float*)d_in[20];

    char* wsp = (char*)d_ws;
    auto alloc = [&](size_t bytes) -> char* {
        char* p = wsp;
        wsp += (bytes + 255) & ~(size_t)255;
        return p;
    };
    unsigned short* Wstack = (unsigned short*)alloc((size_t)4096 * 2048 * 2);
    unsigned short* w1bf   = (unsigned short*)alloc((size_t)128 * 512 * 2);
    unsigned short* w2bf   = (unsigned short*)alloc((size_t)128 * 1024 * 2);
    float* biasstack       = (float*)alloc(4096 * 4);
    float* Cf              = (float*)alloc(128 * 32 * 4);
    unsigned short* enc_bf = (unsigned short*)alloc((size_t)64 * 400 * 512 * 2);
    float* projkey         = (float*)alloc((size_t)64 * 400 * 128 * 4);
    unsigned short* Xbf    = (unsigned short*)alloc((size_t)64 * 2048 * 2);
    float* hbuf            = (float*)alloc((size_t)64 * 1024 * 4);
    float* cbuf            = (float*)alloc((size_t)64 * 1024 * 4);
    float* ctx             = (float*)alloc((size_t)64 * 512 * 4);
    float* pqP             = (float*)alloc((size_t)4 * 64 * 128 * 4);
    float* scorebuf        = (float*)alloc((size_t)64 * 400 * 4);
    float* gatesP          = (float*)alloc((size_t)4 * 64 * 4096 * 4);

    float* mels  = (float*)d_out;                   // [64,600,80]
    float* stops = mels + (size_t)64 * 600 * 80;    // [64,600,1]
    float* attns = stops + (size_t)64 * 600;        // [64,600,400]

    // One-time prep (per launch; deterministic)
    cvt_wstack_kernel<<<(4096 * 2048 + 255) / 256, 256, 0, stream>>>(w_ih, w_hh, Wstack);
    cvt_bf_kernel<<<(128 * 512 + 255) / 256, 256, 0, stream>>>(w1, w1bf, (size_t)128 * 512);
    cvt_bf_kernel<<<(128 * 1024 + 255) / 256, 256, 0, stream>>>(w2, w2bf, (size_t)128 * 1024);
    cvt_bf_kernel<<<((size_t)64 * 400 * 512 + 255) / 256, 256, 0, stream>>>(
        enc, enc_bf, (size_t)64 * 400 * 512);
    bias_kernel<<<16, 256, 0, stream>>>(b_ih, b_hh, biasstack);
    cfilt_kernel<<<16, 256, 0, stream>>>(loc_dense, loc_conv, Cf);
    init_state_kernel<<<(64 * 2048 + 255) / 256, 256, 0, stream>>>(spk, Xbf, hbuf, cbuf);

    // proj_key = enc @ w1^T : [25600,512] x [128,512]^T -> [25600,128] (WMMA)
    gemm_bf16_nt<<<dim3(400, 1, 1), 256, 0, stream>>>(enc_bf, 512, w1bf, 512,
                                                      projkey, 128, 512, nullptr, 0);

    for (int t = 0; t < 600; ++t) {
        prenet_kernel<<<64, 256, 0, stream>>>(teacher, p1w, p1b, p2w, p2b, Xbf, t);
        // pq = h @ w2^T : [64,1024] x [128,1024]^T, split-K x4 -> partials (WMMA)
        gemm_bf16_nt<<<dim3(1, 1, 4), 256, 0, stream>>>(Xbf + 1024, 2048, w2bf, 1024,
                                                        pqP, 128, 256, nullptr,
                                                        (size_t)64 * 128);
        score_kernel<<<dim3(64, 4), 128, 0, stream>>>(projkey, pqP, Cf, v, mask,
                                                      attns, scorebuf, t);
        softmax_ctx_kernel<<<64, 256, 0, stream>>>(scorebuf, enc, attns, ctx, Xbf, t);
        // gates = [x|h] @ [w_ih|w_hh]^T : [64,2048] x [4096,2048]^T, split-K x4 (WMMA)
        gemm_bf16_nt<<<dim3(1, 32, 4), 256, 0, stream>>>(Xbf, 2048, Wstack, 2048,
                                                         gatesP, 4096, 512, nullptr,
                                                         (size_t)64 * 4096);
        lstm_kernel<<<256, 256, 0, stream>>>(gatesP, biasstack, hbuf, cbuf, Xbf);
        output_kernel<<<64, 256, 0, stream>>>(hbuf, ctx, spk, lin_w, lin_b,
                                              stop_w, stop_b, mels, stops, t);
    }
}
